// _MultiHeadAttention_10934986735664
// MI455X (gfx1250) — matrix operand order repaired and checked by reading, not compiled, not run on hardware
//
#include <hip/hip_runtime.h>

typedef _Float16 f16;
typedef __attribute__((ext_vector_type(16))) _Float16 v16h;
typedef __attribute__((ext_vector_type(8)))  _Float16 v8h;
typedef __attribute__((ext_vector_type(8)))  float    v8f;
typedef __attribute__((ext_vector_type(4)))  int      v4i;

#define D_MODEL 512
#define NHEAD   8
#define HD      64
#define SEQ     2048
#define BATCH   4
#define TOKENS  (BATCH * SEQ)     /* 8192 */
#define NQKV    (3 * D_MODEL)     /* 1536 */

__device__ __forceinline__ v8f wmma_f16(v16h a, v16h b, v8f c) {
  // D = A(16x32 f16) * B(32x16 f16) + C(16x16 f32)
  return __builtin_amdgcn_wmma_f32_16x16x32_f16(false, a, false, b, (short)0, c,
                                                false, false);
}

// A-fragment (16x32, f16, row-major source):
// lane L: M = L&15 ; halves 0..7  -> K = kb + (L>>4)*8 + 0..7
//                    halves 8..15 -> K = kb + 16 + (L>>4)*8 + 0..7
__device__ __forceinline__ v16h load_a_frag(const f16* a, int row_stride,
                                            int m_base, int k_base, int lane) {
  const f16* p = a + (size_t)(m_base + (lane & 15)) * row_stride + k_base +
                 ((lane >> 4) << 3);
  union { v16h v; v8h h[2]; } u;
  u.h[0] = *(const v8h*)p;
  u.h[1] = *(const v8h*)(p + 16);
  return u.v;
}

// B-fragment (32x16, f16). Source is B^T stored row-major: bt[n][k].
// lane L: N = L&15 ; halves 0..7  -> K = kb + (L>>4)*8 + 0..7
//                    halves 8..15 -> K = kb + 16 + (L>>4)*8 + 0..7
__device__ __forceinline__ v16h load_b_frag(const f16* bt, int row_stride,
                                            int n_base, int k_base, int lane) {
  const f16* p = bt + (size_t)(n_base + (lane & 15)) * row_stride + k_base +
                 ((lane >> 4) << 3);
  union { v16h v; v8h h[2]; } u;
  u.h[0] = *(const v8h*)p;
  u.h[1] = *(const v8h*)(p + 16);
  return u.v;
}

// 16-byte global -> LDS staging; async copy (ASYNCcnt-tracked, no VGPR
// round-trip) on gfx1250 when the builtin exists, plain copy otherwise.
__device__ __forceinline__ void stage16(const f16* g, f16* l) {
#if __has_builtin(__builtin_amdgcn_global_load_async_to_lds_b128)
  __builtin_amdgcn_global_load_async_to_lds_b128(
      (__attribute__((address_space(1))) v4i*)g,
      (__attribute__((address_space(3))) v4i*)l, 0, 0);
#else
  *(v8h*)l = *(const v8h*)g;
#endif
}

__device__ __forceinline__ void wait_stage() {
#if __has_builtin(__builtin_amdgcn_global_load_async_to_lds_b128) && \
    __has_builtin(__builtin_amdgcn_s_wait_asynccnt)
  __builtin_amdgcn_s_wait_asynccnt(0);
#endif
}

// ---------------------------------------------------------------- converts
__global__ void cvt_f32_to_f16_kernel(const float* __restrict__ src,
                                      f16* __restrict__ dst, int n) {
  int i = blockIdx.x * blockDim.x + threadIdx.x;
  if (i < n) dst[i] = (f16)src[i];
}

// src: [K][N] f32 row-major  ->  dst: [N][K] f16 row-major (i.e. B^T)
__global__ void transpose_to_f16_kernel(const float* __restrict__ src,
                                        f16* __restrict__ dst, int K, int N) {
  int i = blockIdx.x * blockDim.x + threadIdx.x;
  if (i < K * N) {
    int n = i / K;
    int k = i - n * K;
    dst[i] = (f16)src[(size_t)k * N + n];
  }
}

// ------------------------------------------------------- GEMM1: X @ W_qkv
// Writes Q,K as [b,h,t,64] f16 and V transposed as [b,h,64,t] f16.
__global__ __launch_bounds__(256) void qkv_gemm_kernel(
    const f16* __restrict__ Xh, const f16* __restrict__ WqkvT,
    const float* __restrict__ b_qkv, f16* __restrict__ Qb,
    f16* __restrict__ Kb, f16* __restrict__ Vt) {
  const int lane = threadIdx.x & 31;
  const int wave = threadIdx.x >> 5;
  const int m_base = (blockIdx.x * 8 + wave) * 16;
  const int n_strip = blockIdx.y * 64;

  v8f acc[4];
#pragma unroll
  for (int j = 0; j < 4; ++j) {
    float bv = b_qkv[n_strip + j * 16 + (lane & 15)];
#pragma unroll
    for (int r = 0; r < 8; ++r) acc[j][r] = bv;
  }
  for (int kb = 0; kb < D_MODEL; kb += 32) {
    if (kb + 32 < D_MODEL)
      __builtin_prefetch(
          Xh + (size_t)(m_base + (lane & 15)) * D_MODEL + kb + 32, 0, 0);
    v16h a = load_a_frag(Xh, D_MODEL, m_base, kb, lane);
#pragma unroll
    for (int j = 0; j < 4; ++j) {
      v16h b = load_b_frag(WqkvT, D_MODEL, n_strip + j * 16, kb, lane);
      acc[j] = wmma_f16(a, b, acc[j]);
    }
  }
  const int half8 = (lane >> 4) << 3;
#pragma unroll
  for (int j = 0; j < 4; ++j) {
    int cn = n_strip + j * 16 + (lane & 15);
    int which = cn / D_MODEL;              // 0=Q 1=K 2=V (wave-uniform)
    int col = cn - which * D_MODEL;
    int h = col >> 6, d = col & 63;
#pragma unroll
    for (int r = 0; r < 8; ++r) {
      int tok = m_base + half8 + r;
      int b_ = tok >> 11, t = tok & (SEQ - 1);
      size_t bh = (size_t)(b_ * NHEAD + h);
      f16 val = (f16)acc[j][r];
      if (which == 0)
        Qb[(bh * SEQ + t) * HD + d] = val;
      else if (which == 1)
        Kb[(bh * SEQ + t) * HD + d] = val;
      else
        Vt[(bh * HD + d) * SEQ + t] = val;
    }
  }
}

// ---------------------------------------------------- flash attention
// 8 waves/block share one (b,h); K and V^T j-tiles are staged to LDS once
// per block (double buffered, async copies), then each wave computes its own
// 16-query tile.  S^T = K*Q^T so P lands directly in A-fragment layout.
__global__ __launch_bounds__(256) void attn_kernel(
    const f16* __restrict__ Qb, const f16* __restrict__ Kb,
    const f16* __restrict__ Vt, f16* __restrict__ Ah) {
  __shared__ __align__(32) f16 kbuf[2][32 * HD];  // K rows [32][64]
  __shared__ __align__(32) f16 vbuf[2][HD * 32];  // V^T    [64][32]

  const int tid = threadIdx.x;
  const int lane = tid & 31;
  const int wave = tid >> 5;
  const int bh = blockIdx.x >> 4;                 // all waves: same (b,h)
  const int qt = ((blockIdx.x & 15) << 3) + wave; // per-wave q tile
  const int half8 = (lane >> 4) << 3;

  const f16* Q = Qb + (size_t)bh * SEQ * HD + (size_t)qt * 16 * HD;
  const f16* K = Kb + (size_t)bh * SEQ * HD;
  const f16* V = Vt + (size_t)bh * HD * SEQ;

  // cooperative staging coordinates (256 threads x 16B = 4KB per tile)
  const int krow = tid >> 3, kch = (tid & 7) << 3;  // K tile [32][64]
  const int vrow = tid >> 2, vch = (tid & 3) << 3;  // V tile [64][32]

  // Q^T as B fragments (n = q row, k = d), two 32-wide d-steps, kept live.
  v16h qb0 = load_b_frag(Q, HD, 0, 0, lane);
  v16h qb1 = load_b_frag(Q, HD, 0, 32, lane);

  v8f o[4];
#pragma unroll
  for (int dt = 0; dt < 4; ++dt)
#pragma unroll
    for (int r = 0; r < 8; ++r) o[dt][r] = 0.f;

  float m_run = -1e30f, l_run = 0.f;
  const float scale = 0.125f;  // 1/sqrt(64)

  // prologue: stage first j-tile into buffer 0
  stage16(K + (size_t)krow * HD + kch, &kbuf[0][krow * HD + kch]);
  stage16(V + (size_t)vrow * SEQ + vch, &vbuf[0][vrow * 32 + vch]);

  int p = 0;
  for (int jb = 0; jb < SEQ; jb += 32) {
    wait_stage();       // my async copies landed in LDS
    __syncthreads();    // everyone's did
    if (jb + 32 < SEQ) {
      stage16(K + (size_t)(jb + 32 + krow) * HD + kch,
              &kbuf[p ^ 1][krow * HD + kch]);
      stage16(V + (size_t)vrow * SEQ + (jb + 32) + vch,
              &vbuf[p ^ 1][vrow * 32 + vch]);
    }
    const f16* kt = kbuf[p];
    const f16* vt = vbuf[p];

    // K rows as A fragments (from LDS): j-tiles {0,1} x d-steps {0,32}
    v16h a00 = load_a_frag(kt, HD, 0, 0, lane);
    v16h a01 = load_a_frag(kt, HD, 0, 32, lane);
    v16h a10 = load_a_frag(kt, HD, 16, 0, lane);
    v16h a11 = load_a_frag(kt, HD, 16, 32, lane);

    v8f s0, s1;
#pragma unroll
    for (int r = 0; r < 8; ++r) { s0[r] = 0.f; s1[r] = 0.f; }
    s0 = wmma_f16(a00, qb0, s0);
    s0 = wmma_f16(a01, qb1, s0);
    s1 = wmma_f16(a10, qb0, s1);
    s1 = wmma_f16(a11, qb1, s1);

    // per-lane row = q = lane&15 (both halves); online softmax
    float mt = -1e30f;
#pragma unroll
    for (int r = 0; r < 8; ++r) {
      s0[r] *= scale;
      s1[r] *= scale;
      mt = fmaxf(mt, fmaxf(s0[r], s1[r]));
    }
    mt = fmaxf(mt, __shfl_xor(mt, 16, 32));
    float m_new = fmaxf(m_run, mt);
    float alpha = __expf(m_run - m_new);
    m_run = m_new;

    union { v16h v; v8h h[2]; } pf;
    float rsum = 0.f;
#pragma unroll
    for (int r = 0; r < 8; ++r) {
      float p0 = __expf(s0[r] - m_new);
      float p1 = __expf(s1[r] - m_new);
      rsum += p0 + p1;
      pf.h[0][r] = (f16)p0;  // A halves 0..7  <- S^T tile j+0..15
      pf.h[1][r] = (f16)p1;  // A halves 8..15 <- S^T tile j+16..31
    }
    rsum += __shfl_xor(rsum, 16, 32);
    l_run = l_run * alpha + rsum;

    // rescale O: alpha lives in lane q; O rows are VGPR-indexed
#pragma unroll
    for (int r = 0; r < 8; ++r) {
      float aB = __shfl(alpha, half8 + r, 32);
      o[0][r] *= aB; o[1][r] *= aB; o[2][r] *= aB; o[3][r] *= aB;
    }

    // O += P * V   (B fragments from LDS V^T: n = d column, k = j)
#pragma unroll
    for (int dt = 0; dt < 4; ++dt) {
      v16h vbfr = load_b_frag(vt, 32, dt * 16, 0, lane);
      o[dt] = wmma_f16(pf.v, vbfr, o[dt]);
    }
    p ^= 1;
  }

  // normalize and scatter to [b, t, h*64+d] as f16 for the FC GEMM
  const int b_ = bh >> 3, h = bh & 7;
#pragma unroll
  for (int r = 0; r < 8; ++r) {
    float lB = __shfl(l_run, half8 + r, 32);
    float inv = 1.0f / lB;
    int t = qt * 16 + half8 + r;
    size_t rowoff = ((size_t)(b_ * SEQ + t)) * D_MODEL + h * HD;
#pragma unroll
    for (int dt = 0; dt < 4; ++dt)
      Ah[rowoff + dt * 16 + (lane & 15)] = (f16)(o[dt][r] * inv);
  }
}

// ------------------------------------------------- GEMM2: attn @ W_fc -> f32
__global__ __launch_bounds__(256) void fc_gemm_kernel(
    const f16* __restrict__ Ah, const f16* __restrict__ WfcT,
    const float* __restrict__ b_fc, float* __restrict__ out) {
  const int lane = threadIdx.x & 31;
  const int wave = threadIdx.x >> 5;
  const int m_base = (blockIdx.x * 8 + wave) * 16;
  const int n_strip = blockIdx.y * 64;

  v8f acc[4];
#pragma unroll
  for (int j = 0; j < 4; ++j) {
    float bv = b_fc[n_strip + j * 16 + (lane & 15)];
#pragma unroll
    for (int r = 0; r < 8; ++r) acc[j][r] = bv;
  }
  for (int kb = 0; kb < D_MODEL; kb += 32) {
    if (kb + 32 < D_MODEL)
      __builtin_prefetch(
          Ah + (size_t)(m_base + (lane & 15)) * D_MODEL + kb + 32, 0, 0);
    v16h a = load_a_frag(Ah, D_MODEL, m_base, kb, lane);
#pragma unroll
    for (int j = 0; j < 4; ++j) {
      v16h b = load_b_frag(WfcT, D_MODEL, n_strip + j * 16, kb, lane);
      acc[j] = wmma_f16(a, b, acc[j]);
    }
  }
  const int half8 = (lane >> 4) << 3;
#pragma unroll
  for (int j = 0; j < 4; ++j) {
    int n = n_strip + j * 16 + (lane & 15);
#pragma unroll
    for (int r = 0; r < 8; ++r) {
      int m = m_base + half8 + r;
      out[(size_t)m * D_MODEL + n] = acc[j][r];
    }
  }
}

// ---------------------------------------------------------------- launcher
extern "C" void kernel_launch(void* const* d_in, const int* in_sizes, int n_in,
                              void* d_out, int out_size, void* d_ws,
                              size_t ws_size, hipStream_t stream) {
  const float* x     = (const float*)d_in[0];   // [4,2048,512]
  const float* W_qkv = (const float*)d_in[1];   // [512,1536]
  const float* b_qkv = (const float*)d_in[2];   // [1536]
  const float* W_fc  = (const float*)d_in[3];   // [512,512]
  const float* b_fc  = (const float*)d_in[4];   // [512]
  float* out = (float*)d_out;                   // [8192,512]

  char* ws = (char*)d_ws;
  f16* Xh    = (f16*)(ws + 0);                  // 8192*512*2  = 8388608
  f16* WqkvT = (f16*)(ws + 8388608);            // 1536*512*2  = 1572864
  f16* WfcT  = (f16*)(ws + 9961472);            // 512*512*2   = 524288
  f16* Qb    = (f16*)(ws + 10485760);           // 32*2048*64*2= 8388608
  f16* Kb    = (f16*)(ws + 18874368);           // 8388608
  f16* Vt    = (f16*)(ws + 27262976);           // 8388608 ([b,h,64,t])
  f16* Ah    = (f16*)(ws + 35651584);           // 8388608 ([tok,512])

  {
    int n = TOKENS * D_MODEL;  // 4194304
    cvt_f32_to_f16_kernel<<<(n + 255) / 256, 256, 0, stream>>>(x, Xh, n);
  }
  {
    int n = D_MODEL * NQKV;  // dst [1536][512]
    transpose_to_f16_kernel<<<(n + 255) / 256, 256, 0, stream>>>(W_qkv, WqkvT,
                                                                 D_MODEL, NQKV);
  }
  {
    int n = D_MODEL * D_MODEL;  // dst [512][512]
    transpose_to_f16_kernel<<<(n + 255) / 256, 256, 0, stream>>>(W_fc, WfcT,
                                                                 D_MODEL, D_MODEL);
  }

  // QKV projection: M=8192, N=1536, K=512 ; 16x64 per wave, 8 waves/block
  qkv_gemm_kernel<<<dim3(TOKENS / 128, NQKV / 64), 256, 0, stream>>>(
      Xh, WqkvT, b_qkv, Qb, Kb, Vt);

  // attention: 32 bh x 16 blocks, 8 q-tiles (waves) per block
  attn_kernel<<<512, 256, 0, stream>>>(Qb, Kb, Vt, Ah);

  // output projection: M=8192, N=512, K=512
  fc_gemm_kernel<<<dim3(TOKENS / 128, D_MODEL / 64), 256, 0, stream>>>(
      Ah, WfcT, b_fc, out);
}
